// DoubleStreamBlock_41274635714821
// MI455X (gfx1250) — matrix operand order repaired and checked by reading, not compiled, not run on hardware
//
#include <hip/hip_runtime.h>
#include <hip/hip_bf16.h>
#include <cstdint>
#include <cstddef>

#define HS_   1536
#define HEADS_  12
#define HD_    128
#define MLP_  6144
#define LI_   2048
#define LT_    256
#define SEQ_  2304

typedef __attribute__((ext_vector_type(16))) __bf16 v16bf;
typedef __attribute__((ext_vector_type(8)))  float  v8f;

union FragU { v16bf v; uint4 u[2]; };

__device__ __forceinline__ v8f wmma_bf16(v16bf a, v16bf b, v8f c) {
  return __builtin_amdgcn_wmma_f32_16x16x32_bf16(false, a, false, b, (short)0, c, false, false);
}

// A-fragment (16x32 bf16): lane r=lane&15 holds row r; lo-half lanes take K 0..15,
// hi-half K 16..31  -> two contiguous 16B LDS reads.
__device__ __forceinline__ v16bf ld_afrag(const __bf16* s, int stride, int row, int koff, int hi) {
  FragU f;
  f.u[0] = *reinterpret_cast<const uint4*>(s + row * stride + koff + hi * 16);
  f.u[1] = *reinterpret_cast<const uint4*>(s + row * stride + koff + hi * 16 + 8);
  return f.v;
}
// B-fragment (32x16 bf16), operand staged as [N][K]: lane r holds column n=r,
// lo-half K 0..15, hi-half K 16..31 -> one contiguous 32B LDS read.
__device__ __forceinline__ v16bf ld_bfrag(const __bf16* s, int stride, int row, int koff, int hi) {
  const uint4* p = reinterpret_cast<const uint4*>(s + row * stride + koff + hi * 16);
  FragU f; f.u[0] = p[0]; f.u[1] = p[1];
  return f.v;
}

// ---- CDNA5 async global->LDS copy (ASYNCcnt path, VGLOBAL op 98) -------------
// LDS destination address = low 32 bits of the flat shared pointer
__device__ __forceinline__ void async_copy_b128(const void* g, void* l) {
  asm volatile("global_load_async_to_lds_b128 %0, %1, off"
               :: "v"((unsigned)(uintptr_t)l), "v"(g)
               : "memory");
}
__device__ __forceinline__ void wait_async0() {
  asm volatile("s_wait_asynccnt 0x0" ::: "memory");
}

// ---- DPP16 in-row (16-lane) reductions: pure VALU, no LDS traffic ------------
template <int CTRL>
__device__ __forceinline__ float dpp_mov_f32(float v) {
  int i = __float_as_int(v);
  i = __builtin_amdgcn_update_dpp(i, i, CTRL, 0xF, 0xF, true);
  return __int_as_float(i);
}
__device__ __forceinline__ float red16_max(float x) {
  x = fmaxf(x, dpp_mov_f32<0xB1>(x));   // quad_perm [1,0,3,2]
  x = fmaxf(x, dpp_mov_f32<0x4E>(x));   // quad_perm [2,3,0,1]
  x = fmaxf(x, dpp_mov_f32<0x141>(x));  // row_half_mirror
  x = fmaxf(x, dpp_mov_f32<0x140>(x));  // row_mirror
  return x;
}
__device__ __forceinline__ float red16_sum(float x) {
  x = x + dpp_mov_f32<0xB1>(x);
  x = x + dpp_mov_f32<0x4E>(x);
  x = x + dpp_mov_f32<0x141>(x);
  x = x + dpp_mov_f32<0x140>(x);
  return x;
}

__device__ __forceinline__ float gelu_tanh(float x) {
  float x3 = x * x * x;
  return 0.5f * x * (1.0f + tanhf(0.7978845608028654f * (x + 0.044715f * x3)));
}

// ---------------------------------------------------------------------------
// 1) Modulation:  out[j] = b[j] + sum_i silu(vec[i]) * W[i][j]   (9216 cols)
// ---------------------------------------------------------------------------
__global__ __launch_bounds__(256) void modulation_kernel(
    const float* __restrict__ vec, const float* __restrict__ w,
    const float* __restrict__ b, float* __restrict__ out) {
  __shared__ float sv[HS_];
  const int tid = threadIdx.x;
  for (int i = tid; i < HS_; i += 256) {
    float x = vec[i];
    sv[i] = x / (1.0f + __expf(-x));
  }
  __syncthreads();
  const int j = blockIdx.x * 256 + tid;
  float acc = b[j];
  for (int i = 0; i < HS_; ++i) acc += sv[i] * w[(size_t)i * (6 * HS_) + j];
  out[j] = acc;
}

// ---------------------------------------------------------------------------
// 2) LayerNorm + modulation shift/scale -> bf16   (one block per row)
// ---------------------------------------------------------------------------
__global__ __launch_bounds__(256) void ln_mod_kernel(
    const float* __restrict__ x, const float* __restrict__ nsc,
    const float* __restrict__ nb, const float* __restrict__ sc,
    const float* __restrict__ sh, __bf16* __restrict__ out) {
  __shared__ float red[2][8];
  const int row = blockIdx.x, tid = threadIdx.x;
  const int lane = tid & 31, wave = tid >> 5;
  const float* xr = x + (size_t)row * HS_;
  float v[6], s = 0.f, s2 = 0.f;
#pragma unroll
  for (int j = 0; j < 6; ++j) {
    v[j] = xr[tid + j * 256];
    s += v[j]; s2 += v[j] * v[j];
  }
#pragma unroll
  for (int m = 1; m < 32; m <<= 1) { s += __shfl_xor(s, m, 32); s2 += __shfl_xor(s2, m, 32); }
  if (lane == 0) { red[0][wave] = s; red[1][wave] = s2; }
  __syncthreads();
  float ts = 0.f, ts2 = 0.f;
#pragma unroll
  for (int w = 0; w < 8; ++w) { ts += red[0][w]; ts2 += red[1][w]; }
  const float mean = ts * (1.0f / HS_);
  const float var = ts2 * (1.0f / HS_) - mean * mean;
  const float rstd = rsqrtf(var + 1e-6f);
#pragma unroll
  for (int j = 0; j < 6; ++j) {
    int c = tid + j * 256;
    float y = (v[j] - mean) * rstd * nsc[c] + nb[c];
    y = (1.0f + sc[c]) * y + sh[c];
    out[(size_t)row * HS_ + c] = (__bf16)y;
  }
}

// ---------------------------------------------------------------------------
// 3) Transpose + cast weights: src f32 [R][C]  ->  dst bf16 [C][R]
// ---------------------------------------------------------------------------
__global__ __launch_bounds__(256) void transpose_cast_kernel(
    const float* __restrict__ src, __bf16* __restrict__ dst, int R, int C) {
  __shared__ float t[32][33];
  const int rb = blockIdx.y * 32, cb = blockIdx.x * 32;
  const int tr = threadIdx.x >> 5;   // 0..7
  const int tc = threadIdx.x & 31;   // 0..31
#pragma unroll
  for (int i = 0; i < 4; ++i)
    t[tr + i * 8][tc] = src[(size_t)(rb + tr + i * 8) * C + cb + tc];
  __syncthreads();
#pragma unroll
  for (int i = 0; i < 4; ++i)
    dst[(size_t)(cb + tr + i * 8) * R + rb + tc] = (__bf16)t[tc][tr + i * 8];
}

// ---------------------------------------------------------------------------
// 4) WMMA GEMM:  C[M][N] = epi(A[M][K](bf16) x BT[N][K](bf16))
//    Tiles staged with global_load_async_to_lds_b128 (ASYNCcnt double buffer).
//    MODE 0: store raw f32      MODE 1: bias+gelu -> bf16
//    MODE 2: resid + gate*(acc+bias) -> f32
// ---------------------------------------------------------------------------
#define BM 128
#define BN 128
#define BK 32
#define AST 40   // padded LDS row stride (bf16 elements)

template <int MODE>
__global__ __launch_bounds__(256) void gemm_wmma_kernel(
    const __bf16* __restrict__ A, const __bf16* __restrict__ BT,
    int M, int N, int K,
    const float* __restrict__ bias, const float* __restrict__ gate,
    const float* __restrict__ resid, float* __restrict__ outF,
    __bf16* __restrict__ outH) {
  __shared__ __bf16 As[2][BM * AST];
  __shared__ __bf16 Bs[2][BN * AST];

  const int tid = threadIdx.x;
  const int lane = tid & 31, wave = tid >> 5;
  const int wm = wave >> 2, wn = wave & 3;      // 2 x 4 wave grid
  const int r = lane & 15, hi = lane >> 4;
  const int bm = blockIdx.y * BM, bn = blockIdx.x * BN;
  const int lrow = tid >> 2;                    // 0..63
  const int lcol = (tid & 3) * 8;               // 0,8,16,24

  v8f acc[4][2];
  const v8f vzero = {0.f, 0.f, 0.f, 0.f, 0.f, 0.f, 0.f, 0.f};
#pragma unroll
  for (int mt = 0; mt < 4; ++mt)
#pragma unroll
    for (int nt = 0; nt < 2; ++nt) acc[mt][nt] = vzero;

  auto issue_async = [&](int buf, int kt) {
    async_copy_b128(A + (size_t)(bm + lrow) * K + kt * BK + lcol,
                    &As[buf][lrow * AST + lcol]);
    async_copy_b128(A + (size_t)(bm + lrow + 64) * K + kt * BK + lcol,
                    &As[buf][(lrow + 64) * AST + lcol]);
    async_copy_b128(BT + (size_t)(bn + lrow) * K + kt * BK + lcol,
                    &Bs[buf][lrow * AST + lcol]);
    async_copy_b128(BT + (size_t)(bn + lrow + 64) * K + kt * BK + lcol,
                    &Bs[buf][(lrow + 64) * AST + lcol]);
  };

  const int nk = K / BK;
  issue_async(0, 0);
  wait_async0();
  __syncthreads();

  for (int kt = 0; kt < nk; ++kt) {
    const int cur = kt & 1;
    if (kt + 1 < nk) issue_async(cur ^ 1, kt + 1);  // overlap with WMMA below
    v16bf af[4], bf[2];
#pragma unroll
    for (int mt = 0; mt < 4; ++mt)
      af[mt] = ld_afrag(As[cur], AST, wm * 64 + mt * 16 + r, 0, hi);
#pragma unroll
    for (int nt = 0; nt < 2; ++nt)
      bf[nt] = ld_bfrag(Bs[cur], AST, wn * 32 + nt * 16 + r, 0, hi);
#pragma unroll
    for (int mt = 0; mt < 4; ++mt)
#pragma unroll
      for (int nt = 0; nt < 2; ++nt)
        acc[mt][nt] = wmma_bf16(af[mt], bf[nt], acc[mt][nt]);
    if (kt + 1 < nk) wait_async0();
    __syncthreads();
  }

#pragma unroll
  for (int mt = 0; mt < 4; ++mt)
#pragma unroll
    for (int nt = 0; nt < 2; ++nt) {
      const int gc = bn + wn * 32 + nt * 16 + r;
      float bv = 0.f, gv = 0.f;
      if constexpr (MODE >= 1) bv = bias[gc];
      if constexpr (MODE == 2) gv = gate[gc];
#pragma unroll
      for (int j = 0; j < 8; ++j) {
        const int gr = bm + wm * 64 + mt * 16 + j + hi * 8;
        float v = acc[mt][nt][j];
        if constexpr (MODE == 0) {
          outF[(size_t)gr * N + gc] = v;
        } else if constexpr (MODE == 1) {
          outH[(size_t)gr * N + gc] = (__bf16)gelu_tanh(v + bv);
        } else {
          outF[(size_t)gr * N + gc] = resid[(size_t)gr * N + gc] + gv * (v + bv);
        }
      }
    }
}

// ---------------------------------------------------------------------------
// 5) RMS-norm(q,k) + RoPE; emit q,k as [h][l][d] bf16 and v transposed [h][d][l]
//    One warp per (h,l) pair; lane owns 4 of the 128 dims.
// ---------------------------------------------------------------------------
__global__ __launch_bounds__(256) void rmsnorm_rope_kernel(
    const float* __restrict__ qkv_i, const float* __restrict__ qkv_t,
    const float* __restrict__ pe,
    const float* __restrict__ iqs, const float* __restrict__ iks,
    const float* __restrict__ tqs, const float* __restrict__ tks,
    __bf16* __restrict__ qh, __bf16* __restrict__ kh, __bf16* __restrict__ vt) {
  const int pair = blockIdx.x * 8 + (threadIdx.x >> 5);
  const int lane = threadIdx.x & 31;
  const int l = pair % SEQ_;
  const int h = pair / SEQ_;
  const float* src;
  const float* qs;
  const float* ks;
  if (l < LT_) { src = qkv_t + (size_t)l * (3 * HS_); qs = tqs; ks = tks; }
  else         { src = qkv_i + (size_t)(l - LT_) * (3 * HS_); qs = iqs; ks = iks; }
  const int d0 = lane * 4;
  float q[4], k[4], v[4];
#pragma unroll
  for (int j = 0; j < 4; ++j) {
    q[j] = src[h * HD_ + d0 + j];
    k[j] = src[HS_ + h * HD_ + d0 + j];
    v[j] = src[2 * HS_ + h * HD_ + d0 + j];
  }
  float sq = 0.f, sk = 0.f;
#pragma unroll
  for (int j = 0; j < 4; ++j) { sq += q[j] * q[j]; sk += k[j] * k[j]; }
#pragma unroll
  for (int m = 1; m < 32; m <<= 1) { sq += __shfl_xor(sq, m, 32); sk += __shfl_xor(sk, m, 32); }
  const float rq = rsqrtf(sq * (1.0f / HD_) + 1e-6f);
  const float rk = rsqrtf(sk * (1.0f / HD_) + 1e-6f);
#pragma unroll
  for (int j = 0; j < 4; ++j) { q[j] *= rq * qs[d0 + j]; k[j] *= rk * ks[d0 + j]; }
#pragma unroll
  for (int p = 0; p < 2; ++p) {
    const int i = lane * 2 + p;  // half-dim index 0..63
    const float* pp = pe + ((size_t)l * 64 + i) * 4;
    const float c0 = pp[0], c1 = pp[1], c2 = pp[2], c3 = pp[3];
    const float q0 = q[2 * p], q1 = q[2 * p + 1];
    const float k0 = k[2 * p], k1 = k[2 * p + 1];
    const size_t base = ((size_t)h * SEQ_ + l) * HD_ + d0 + 2 * p;
    qh[base]     = (__bf16)(c0 * q0 + c1 * q1);
    qh[base + 1] = (__bf16)(c2 * q0 + c3 * q1);
    kh[base]     = (__bf16)(c0 * k0 + c1 * k1);
    kh[base + 1] = (__bf16)(c2 * k0 + c3 * k1);
  }
#pragma unroll
  for (int j = 0; j < 4; ++j)
    vt[((size_t)h * HD_ + d0 + j) * SEQ_ + l] = (__bf16)v[j];
}

// ---------------------------------------------------------------------------
// 6) Flash attention (WMMA, online softmax). Grid (qblocks=18, heads=12),
//    8 waves x 16 q-rows = 128 q-rows per block, K/V blocks of 32.
//    Tiles staged via async global->LDS; row reductions via DPP16.
// ---------------------------------------------------------------------------
__global__ __launch_bounds__(256) void flash_attn_kernel(
    const __bf16* __restrict__ Q, const __bf16* __restrict__ Kb,
    const __bf16* __restrict__ VT, __bf16* __restrict__ Out) {
  constexpr int KBLK = 32;
  constexpr float SCALE = 0.08838834764831845f;  // 1/sqrt(128)
  __shared__ __bf16 Qs[128 * 136];
  __shared__ __bf16 Ks[KBLK * 136];
  __shared__ __bf16 VTs[128 * 40];
  __shared__ __bf16 Ps[128 * 40];

  const int h = blockIdx.y;
  const int qb = blockIdx.x * 128;
  const int tid = threadIdx.x, lane = tid & 31, wave = tid >> 5;
  const int r = lane & 15, hi = lane >> 4;
  const __bf16* Qh = Q + (size_t)h * SEQ_ * HD_;
  const __bf16* Kh = Kb + (size_t)h * SEQ_ * HD_;
  const __bf16* Vh = VT + (size_t)h * HD_ * SEQ_;

  for (int i = tid; i < 128 * 16; i += 256) {
    const int row = i >> 4, c = (i & 15) * 8;
    async_copy_b128(Qh + (size_t)(qb + row) * HD_ + c, Qs + row * 136 + c);
  }
  float m_[8], l_[8];
  v8f o[8];
  const v8f vzero = {0.f, 0.f, 0.f, 0.f, 0.f, 0.f, 0.f, 0.f};
#pragma unroll
  for (int j = 0; j < 8; ++j) { m_[j] = -3.0e38f; l_[j] = 0.f; }
#pragma unroll
  for (int nt = 0; nt < 8; ++nt) o[nt] = vzero;
  wait_async0();
  __syncthreads();

  for (int kb = 0; kb < SEQ_ / KBLK; ++kb) {
    const int k0 = kb * KBLK;
    for (int i = tid; i < KBLK * 16; i += 256) {
      const int row = i >> 4, c = (i & 15) * 8;
      async_copy_b128(Kh + (size_t)(k0 + row) * HD_ + c, Ks + row * 136 + c);
    }
    for (int i = tid; i < 128 * 4; i += 256) {
      const int d = i >> 2, c = (i & 3) * 8;
      async_copy_b128(Vh + (size_t)d * SEQ_ + k0 + c, VTs + d * 40 + c);
    }
    wait_async0();
    __syncthreads();

    // S = Q K^T  (16 rows x 32 keys per wave)
    v8f s[2];
    s[0] = vzero; s[1] = vzero;
#pragma unroll
    for (int kt = 0; kt < 4; ++kt) {
      const v16bf aq = ld_afrag(Qs, 136, wave * 16 + r, kt * 32, hi);
#pragma unroll
      for (int nt = 0; nt < 2; ++nt) {
        const v16bf bk = ld_bfrag(Ks, 136, nt * 16 + r, kt * 32, hi);
        s[nt] = wmma_bf16(aq, bk, s[nt]);
      }
    }
    // online softmax; row j lives in VGPR slot j, lo/hi half-wave = rows j / j+8
#pragma unroll
    for (int j = 0; j < 8; ++j) {
      const float s0 = s[0][j] * SCALE;
      const float s1 = s[1][j] * SCALE;
      const float mx = red16_max(fmaxf(s0, s1));
      const float nm = fmaxf(m_[j], mx);
      const float alpha = __expf(m_[j] - nm);
      const float p0 = __expf(s0 - nm);
      const float p1 = __expf(s1 - nm);
      const float rs = red16_sum(p0 + p1);
      m_[j] = nm;
      l_[j] = l_[j] * alpha + rs;
#pragma unroll
      for (int nt = 0; nt < 8; ++nt) o[nt][j] = o[nt][j] * alpha;
      const int prow = wave * 16 + j + hi * 8;
      Ps[prow * 40 + r] = (__bf16)p0;
      Ps[prow * 40 + 16 + r] = (__bf16)p1;
    }
    // O += P V   (P rows are wave-local; LDS ops in-order within a wave)
    const v16bf ap = ld_afrag(Ps, 40, wave * 16 + r, 0, hi);
#pragma unroll
    for (int nt = 0; nt < 8; ++nt) {
      const v16bf bv = ld_bfrag(VTs, 40, nt * 16 + r, 0, hi);
      o[nt] = wmma_bf16(ap, bv, o[nt]);
    }
    __syncthreads();
  }

#pragma unroll
  for (int j = 0; j < 8; ++j) {
    const float inv = 1.0f / l_[j];
    const int gr = qb + wave * 16 + j + hi * 8;
#pragma unroll
    for (int nt = 0; nt < 8; ++nt)
      Out[(size_t)gr * HS_ + h * HD_ + nt * 16 + r] = (__bf16)(o[nt][j] * inv);
  }
}

// ---------------------------------------------------------------------------
extern "C" void kernel_launch(void* const* d_in, const int* in_sizes, int n_in,
                              void* d_out, int out_size, void* d_ws, size_t ws_size,
                              hipStream_t stream) {
  const float* img   = (const float*)d_in[0];
  const float* txt   = (const float*)d_in[1];
  const float* vec   = (const float*)d_in[2];
  const float* pe    = (const float*)d_in[3];
  const float* img_mod_w = (const float*)d_in[4];
  const float* img_mod_b = (const float*)d_in[5];
  const float* txt_mod_w = (const float*)d_in[6];
  const float* txt_mod_b = (const float*)d_in[7];
  const float* img_n1s = (const float*)d_in[8];
  const float* img_n1b = (const float*)d_in[9];
  const float* txt_n1s = (const float*)d_in[10];
  const float* txt_n1b = (const float*)d_in[11];
  const float* img_qkv_w = (const float*)d_in[12];
  const float* txt_qkv_w = (const float*)d_in[13];
  const float* img_qs = (const float*)d_in[14];
  const float* img_ks = (const float*)d_in[15];
  const float* txt_qs = (const float*)d_in[16];
  const float* txt_ks = (const float*)d_in[17];
  const float* img_proj_w = (const float*)d_in[18];
  const float* img_proj_b = (const float*)d_in[19];
  const float* txt_proj_w = (const float*)d_in[20];
  const float* txt_proj_b = (const float*)d_in[21];
  const float* img_n2s = (const float*)d_in[22];
  const float* img_n2b = (const float*)d_in[23];
  const float* txt_n2s = (const float*)d_in[24];
  const float* txt_n2b = (const float*)d_in[25];
  const float* img_mlp_w1 = (const float*)d_in[26];
  const float* img_mlp_b1 = (const float*)d_in[27];
  const float* img_mlp_w2 = (const float*)d_in[28];
  const float* img_mlp_b2 = (const float*)d_in[29];
  const float* txt_mlp_w1 = (const float*)d_in[30];
  const float* txt_mlp_b1 = (const float*)d_in[31];
  const float* txt_mlp_w2 = (const float*)d_in[32];
  const float* txt_mlp_b2 = (const float*)d_in[33];

  char* wsp = (char*)d_ws;
  auto bump = [&](size_t bytes) -> void* {
    void* p = wsp;
    wsp += (bytes + 255) & ~(size_t)255;
    return p;
  };
  float*  mod     = (float*)bump((size_t)12 * HS_ * 4);      // [img sh1,sc1,g1,sh2,sc2,g2][txt ...]
  __bf16* img_m   = (__bf16*)bump((size_t)LI_ * HS_ * 2);    // reused later as h_img
  __bf16* txt_m   = (__bf16*)bump((size_t)LT_ * HS_ * 2);    // reused later as h_txt
  __bf16* qkvT_i  = (__bf16*)bump((size_t)3 * HS_ * HS_ * 2);
  __bf16* qkvT_t  = (__bf16*)bump((size_t)3 * HS_ * HS_ * 2);
  __bf16* projT_i = (__bf16*)bump((size_t)HS_ * HS_ * 2);
  __bf16* projT_t = (__bf16*)bump((size_t)HS_ * HS_ * 2);
  __bf16* mlp1T_i = (__bf16*)bump((size_t)MLP_ * HS_ * 2);
  __bf16* mlp1T_t = (__bf16*)bump((size_t)MLP_ * HS_ * 2);
  __bf16* mlp2T_i = (__bf16*)bump((size_t)HS_ * MLP_ * 2);
  __bf16* mlp2T_t = (__bf16*)bump((size_t)HS_ * MLP_ * 2);
  float*  qkv_i   = (float*)bump((size_t)LI_ * 3 * HS_ * 4);
  float*  qkv_t   = (float*)bump((size_t)LT_ * 3 * HS_ * 4);
  __bf16* qh      = (__bf16*)bump((size_t)HEADS_ * SEQ_ * HD_ * 2);
  __bf16* kh      = (__bf16*)bump((size_t)HEADS_ * SEQ_ * HD_ * 2);
  __bf16* vt      = (__bf16*)bump((size_t)HEADS_ * HD_ * SEQ_ * 2);
  __bf16* attn    = (__bf16*)bump((size_t)SEQ_ * HS_ * 2);
  float*  img2    = (float*)bump((size_t)LI_ * HS_ * 4);
  float*  txt2    = (float*)bump((size_t)LT_ * HS_ * 4);
  __bf16* mlph_i  = (__bf16*)qkv_i;  // reuse (25.2MB <= 37.7MB), qkv dead after rope
  __bf16* mlph_t  = (__bf16*)qkv_t;  // reuse ( 3.1MB <=  4.7MB)

  float* out_img = (float*)d_out;
  float* out_txt = out_img + (size_t)LI_ * HS_;

  // 1) modulation vectors
  modulation_kernel<<<36, 256, 0, stream>>>(vec, img_mod_w, img_mod_b, mod);
  modulation_kernel<<<36, 256, 0, stream>>>(vec, txt_mod_w, txt_mod_b, mod + 6 * HS_);

  // 2) norm1 + mod
  ln_mod_kernel<<<LI_, 256, 0, stream>>>(img, img_n1s, img_n1b, mod + 1 * HS_, mod + 0 * HS_, img_m);
  ln_mod_kernel<<<LT_, 256, 0, stream>>>(txt, txt_n1s, txt_n1b, mod + 7 * HS_, mod + 6 * HS_, txt_m);

  // 3) weight transpose+cast to bf16 [N][K]
  transpose_cast_kernel<<<dim3(3 * HS_ / 32, HS_ / 32), 256, 0, stream>>>(img_qkv_w, qkvT_i, HS_, 3 * HS_);
  transpose_cast_kernel<<<dim3(3 * HS_ / 32, HS_ / 32), 256, 0, stream>>>(txt_qkv_w, qkvT_t, HS_, 3 * HS_);
  transpose_cast_kernel<<<dim3(HS_ / 32, HS_ / 32), 256, 0, stream>>>(img_proj_w, projT_i, HS_, HS_);
  transpose_cast_kernel<<<dim3(HS_ / 32, HS_ / 32), 256, 0, stream>>>(txt_proj_w, projT_t, HS_, HS_);
  transpose_cast_kernel<<<dim3(MLP_ / 32, HS_ / 32), 256, 0, stream>>>(img_mlp_w1, mlp1T_i, HS_, MLP_);
  transpose_cast_kernel<<<dim3(MLP_ / 32, HS_ / 32), 256, 0, stream>>>(txt_mlp_w1, mlp1T_t, HS_, MLP_);
  transpose_cast_kernel<<<dim3(HS_ / 32, MLP_ / 32), 256, 0, stream>>>(img_mlp_w2, mlp2T_i, MLP_, HS_);
  transpose_cast_kernel<<<dim3(HS_ / 32, MLP_ / 32), 256, 0, stream>>>(txt_mlp_w2, mlp2T_t, MLP_, HS_);

  // 4) QKV GEMMs (raw f32 out)
  gemm_wmma_kernel<0><<<dim3(3 * HS_ / BN, LI_ / BM), 256, 0, stream>>>(
      img_m, qkvT_i, LI_, 3 * HS_, HS_, nullptr, nullptr, nullptr, qkv_i, nullptr);
  gemm_wmma_kernel<0><<<dim3(3 * HS_ / BN, LT_ / BM), 256, 0, stream>>>(
      txt_m, qkvT_t, LT_, 3 * HS_, HS_, nullptr, nullptr, nullptr, qkv_t, nullptr);

  // 5) qk rms-norm + rope; build head-major q/k and transposed v
  rmsnorm_rope_kernel<<<(HEADS_ * SEQ_) / 8, 256, 0, stream>>>(
      qkv_i, qkv_t, pe, img_qs, img_ks, txt_qs, txt_ks, qh, kh, vt);

  // 6) attention
  flash_attn_kernel<<<dim3(SEQ_ / 128, HEADS_), 256, 0, stream>>>(qh, kh, vt, attn);

  // 7) projection + gated residual
  gemm_wmma_kernel<2><<<dim3(HS_ / BN, LI_ / BM), 256, 0, stream>>>(
      attn + (size_t)LT_ * HS_, projT_i, LI_, HS_, HS_,
      img_proj_b, mod + 2 * HS_, img, img2, nullptr);
  gemm_wmma_kernel<2><<<dim3(HS_ / BN, LT_ / BM), 256, 0, stream>>>(
      attn, projT_t, LT_, HS_, HS_,
      txt_proj_b, mod + 8 * HS_, txt, txt2, nullptr);

  // 8) norm2 + mod (reuse img_m/txt_m as h)
  ln_mod_kernel<<<LI_, 256, 0, stream>>>(img2, img_n2s, img_n2b, mod + 4 * HS_, mod + 3 * HS_, img_m);
  ln_mod_kernel<<<LT_, 256, 0, stream>>>(txt2, txt_n2s, txt_n2b, mod + 10 * HS_, mod + 9 * HS_, txt_m);

  // 9) MLP up + gelu -> bf16
  gemm_wmma_kernel<1><<<dim3(MLP_ / BN, LI_ / BM), 256, 0, stream>>>(
      img_m, mlp1T_i, LI_, MLP_, HS_, img_mlp_b1, nullptr, nullptr, nullptr, mlph_i);
  gemm_wmma_kernel<1><<<dim3(MLP_ / BN, LT_ / BM), 256, 0, stream>>>(
      txt_m, mlp1T_t, LT_, MLP_, HS_, txt_mlp_b1, nullptr, nullptr, nullptr, mlph_t);

  // 10) MLP down + gated residual -> d_out
  gemm_wmma_kernel<2><<<dim3(HS_ / BN, LI_ / BM), 256, 0, stream>>>(
      mlph_i, mlp2T_i, LI_, HS_, MLP_, img_mlp_b2, mod + 5 * HS_, img2, out_img, nullptr);
  gemm_wmma_kernel<2><<<dim3(HS_ / BN, LT_ / BM), 256, 0, stream>>>(
      mlph_t, mlp2T_t, LT_, HS_, MLP_, txt_mlp_b2, mod + 11 * HS_, txt2, out_txt, nullptr);
}
